// SemGCN_Camera_Concat_World_50964081935411
// MI455X (gfx1250) — matrix operand order repaired and checked by reading, not compiled, not run on hardware
//
#include <hip/hip_runtime.h>
#include <hip/hip_bf16.h>

typedef __attribute__((ext_vector_type(16))) _Float16 v16h;
typedef __attribute__((ext_vector_type(8)))  _Float16 v8h;
typedef __attribute__((ext_vector_type(8)))  float    v8f;
typedef __attribute__((ext_vector_type(4)))  unsigned int v4u;
typedef __attribute__((ext_vector_type(8)))  int      v8i;
typedef __attribute__((ext_vector_type(4)))  int      v4i;

#define LDSX 136   // padded LDS row stride in halfs (128 + 8) -> 256B row + 16B pad

static __device__ __forceinline__ v16h join16(v8h p, v8h q) {
  return __builtin_shufflevector(p, q, 0,1,2,3,4,5,6,7,8,9,10,11,12,13,14,15);
}

// ---------------------------------------------------------------------------
// Attention prep: att = softmax_row(where(mask, e, -inf)); 15 matrices.
// ---------------------------------------------------------------------------
__global__ void prep_att(const float* __restrict__ e_in, const float* __restrict__ e_cat,
                         const float* __restrict__ e_res, const float* __restrict__ e_out,
                         float* __restrict__ att) {
  int m = blockIdx.x;          // 0..14 : [in, cat, res0..7, out0..4]
  int i = threadIdx.x;         // 0..15 row
  const float* e;
  if (m == 0)       e = e_in;
  else if (m == 1)  e = e_cat;
  else if (m < 10)  e = e_res + (m - 2) * 256;
  else              e = e_out + (m - 10) * 256;

  const int E0[15] = {0,1,2,0,4,5,0,7,8,9,8,11,12,8,14};
  const int E1[15] = {1,2,3,4,5,6,7,8,9,10,11,12,13,14,15};
  unsigned row = 1u << i;
  #pragma unroll
  for (int k = 0; k < 15; ++k) {
    if (E0[k] == i) row |= 1u << E1[k];
    if (E1[k] == i) row |= 1u << E0[k];
  }
  float mx = -1e30f;
  for (int j = 0; j < 16; ++j) if ((row >> j) & 1) mx = fmaxf(mx, e[i*16 + j]);
  float ex[16], s = 0.f;
  for (int j = 0; j < 16; ++j) {
    float v = ((row >> j) & 1) ? __expf(e[i*16 + j] - mx) : 0.f;
    ex[j] = v; s += v;
  }
  float inv = 1.f / s;
  for (int j = 0; j < 16; ++j) att[m*256 + i*16 + j] = ex[j] * inv;
}

// ---------------------------------------------------------------------------
// Convert W_res fp32 [8][2][f=128][o=128] -> f16 transposed [8][2][o][f]
// ---------------------------------------------------------------------------
__global__ __launch_bounds__(256) void convw(const float* __restrict__ W, _Float16* __restrict__ Wt) {
  int gid = blockIdx.x * 256 + threadIdx.x;     // 262144 threads, 1 elem each
  if (gid < 8*2*128*128) {
    int f = gid & 127, o = (gid >> 7) & 127, lm = gid >> 14;
    Wt[gid] = (_Float16)W[lm*16384 + f*128 + o];
  }
}

__global__ void zero_stats(float* stats) { stats[threadIdx.x] = 0.f; }

// ---------------------------------------------------------------------------
// WMMA semgconv, Fin=Fout=128.  Block = 8 batches x 16 joints x 128 ch.
// W tiles staged global->LDS by the Tensor Data Mover (padded rows),
// X tile staged manually (needs fp32->f16 conversion).
// ---------------------------------------------------------------------------
__global__ __launch_bounds__(256)
void gconv128(const float* __restrict__ X, const _Float16* __restrict__ W0g,
              const _Float16* __restrict__ W1g, const float* __restrict__ bias,
              const float* __restrict__ att, float* __restrict__ Y,
              float* __restrict__ stats) {
  extern __shared__ char smem[];
  _Float16* Xs   = (_Float16*)smem;            // 128 x LDSX
  _Float16* Ws0  = Xs  + 128*LDSX;             // [o][f] 128 x LDSX (padded rows)
  _Float16* Ws1  = Ws0 + 128*LDSX;
  float*    Hs   = (float*)(Ws1 + 128*LDSX);   // 8 waves x 2 x 16 x 17
  float*    atts = Hs + 8*2*16*17;
  float*    biass= atts + 256;
  float*    lsum = biass + 128;
  float*    lsq  = lsum + 128;

  const int t    = threadIdx.x;
  const int lane = t & 31;
  const int w    = t >> 5;                     // wave 0..7 -> batch-in-tile
  const int R0   = blockIdx.x * 128;           // global row base (b*16+j)

#if __has_builtin(__builtin_amdgcn_tensor_load_to_lds)
  // --- TDM: DMA both 128x128 f16 weight tiles into LDS with 16B row padding.
  if (w == 0) {
    // D# group1 (256b): data_size=2B, pad_enable, pad_interval=64DW(256B),
    // pad_amount=4DW(16B), tensor 128x128, tile 128x128, dim0 stride 128.
    v8i g1 = { (int)((1u<<16) | (1u<<20) | (5u<<22) | (3u<<25)),
               (int)(128u << 16),   // tensor_dim0[15:0] @ bits[63:48]
               (int)(128u << 16),   // tensor_dim1[15:0] @ bits[95:80]
               (int)(128u << 16),   // tile_dim0        @ bits[127:112]
               128,                 // tile_dim1        @ bits[143:128]
               128,                 // tensor_dim0_stride[31:0] @ bits[191:160]
               0, 0 };
    v4i gz4 = {0, 0, 0, 0};
    v8i gz8 = {0, 0, 0, 0, 0, 0, 0, 0};
    unsigned long long ga = (unsigned long long)(uintptr_t)W0g;
    unsigned long long gb = (unsigned long long)(uintptr_t)W1g;
    v4u g0a = { 1u, (unsigned)(uintptr_t)(void*)Ws0,
                (unsigned)ga, (unsigned)((ga >> 32) & 0x1FFFFFFu) | (2u << 30) };
    v4u g0b = { 1u, (unsigned)(uintptr_t)(void*)Ws1,
                (unsigned)gb, (unsigned)((gb >> 32) & 0x1FFFFFFu) | (2u << 30) };
    __builtin_amdgcn_tensor_load_to_lds(g0a, g1, gz4, gz4, gz8, 0);
    __builtin_amdgcn_tensor_load_to_lds(g0b, g1, gz4, gz4, gz8, 0);
    __builtin_amdgcn_s_wait_tensorcnt(0);
  }
#else
  #pragma unroll
  for (int it = 0; it < 8; ++it) {
    int idx = t + it*256;                      // 2048 uint4 per matrix
    int o = idx >> 4, f8 = idx & 15;
    *(uint4*)(Ws0 + o*LDSX + f8*8) = ((const uint4*)W0g)[idx];
    *(uint4*)(Ws1 + o*LDSX + f8*8) = ((const uint4*)W1g)[idx];
  }
#endif

  // stage X tile (fp32 -> f16)
  #pragma unroll
  for (int it = 0; it < 16; ++it) {
    int idx = t + it*256;                      // 4096 float4 loads
    int row = idx >> 5, c4 = idx & 31;
    float4 v = ((const float4*)X)[(long)(R0 + row)*32 + c4];
    _Float16* d = Xs + row*LDSX + c4*4;
    d[0] = (_Float16)v.x; d[1] = (_Float16)v.y;
    d[2] = (_Float16)v.z; d[3] = (_Float16)v.w;
  }
  if (t < 128) { biass[t] = bias[t]; lsum[t] = 0.f; lsq[t] = 0.f; }
  atts[t] = att[t];
  __syncthreads();

  const int lo = lane & 15, hi = lane >> 4;

  // A fragments for this wave's 16 rows, all K (f16 16x32 layout)
  v16h afr[4];
  {
    const _Float16* xr = Xs + (w*16 + lo)*LDSX;
    #pragma unroll
    for (int kk = 0; kk < 4; ++kk) {
      int base = kk*32 + hi*16;
      afr[kk] = join16(*(const v8h*)(xr + base), *(const v8h*)(xr + base + 8));
    }
  }

  float* hw = Hs + w * (2*16*17);

  for (int n = 0; n < 8; ++n) {
    v8f acc0 = {0,0,0,0,0,0,0,0};
    v8f acc1 = {0,0,0,0,0,0,0,0};
    const _Float16* wr0 = Ws0 + (n*16 + lo)*LDSX;   // B-frag row: N = n*16+lo
    const _Float16* wr1 = Ws1 + (n*16 + lo)*LDSX;
    #pragma unroll
    for (int kk = 0; kk < 4; ++kk) {
      int fb = kk*32 + hi*16;                       // 16 contiguous K per lane
      v16h b0 = join16(*(const v8h*)(wr0 + fb), *(const v8h*)(wr0 + fb + 8));
      v16h b1 = join16(*(const v8h*)(wr1 + fb), *(const v8h*)(wr1 + fb + 8));
      acc0 = __builtin_amdgcn_wmma_f32_16x16x32_f16(false, afr[kk], false, b0,
                                                    (short)0, acc0, false, false);
      acc1 = __builtin_amdgcn_wmma_f32_16x16x32_f16(false, afr[kk], false, b1,
                                                    (short)0, acc1, false, false);
    }
    // spill C tiles to per-wave LDS so we can mix across joints (M dim)
    #pragma unroll
    for (int r = 0; r < 8; ++r) {
      int M = r + 8*hi;
      hw[M*17 + lo]       = acc0[r];
      hw[272 + M*17 + lo] = acc1[r];
    }
    asm volatile("" ::: "memory");   // DS is in-order per wave; block reordering
    float h1v[16];
    #pragma unroll
    for (int j = 0; j < 16; ++j) h1v[j] = hw[272 + j*17 + lo];
    int o = n*16 + lo;
    float bo = biass[o];
    float p1 = 0.f, p2 = 0.f;
    #pragma unroll
    for (int r = 0; r < 8; ++r) {
      int i = r + 8*hi;
      const float* ar = atts + i*16;
      float acc = 0.f;
      #pragma unroll
      for (int j = 0; j < 16; ++j) acc += ar[j] * h1v[j];
      acc += ar[i] * (acc0[r] - h1v[i]);   // diag uses h0 instead of h1
      acc += bo;
      Y[(long)(R0 + w*16 + i)*128 + o] = acc;
      p1 += acc; p2 += acc*acc;
    }
    atomicAdd(lsum + o, p1);
    atomicAdd(lsq  + o, p2);
    asm volatile("" ::: "memory");
  }
  __syncthreads();
  if (t < 128) {
    atomicAdd(stats + t,       lsum[t]);
    atomicAdd(stats + 128 + t, lsq[t]);
  }
}

// ---------------------------------------------------------------------------
// Scalar semgconv for tiny Fin (2 or 12), Fout=128, with BN stats.
// ---------------------------------------------------------------------------
template<int FIN>
__global__ __launch_bounds__(256)
void gconv_in(const float* __restrict__ X, const float* __restrict__ W,
              const float* __restrict__ bias, const float* __restrict__ att,
              float* __restrict__ Y, float* __restrict__ stats) {
  __shared__ float lsum[128], lsq[128], atts[256];
  int t = threadIdx.x;
  if (t < 128) { lsum[t] = 0.f; lsq[t] = 0.f; }
  atts[t] = att[t];
  __syncthreads();
  float s1 = 0.f, s2 = 0.f; int o_me = -1;
  const long N = (long)8192*16*128;
  for (long gid = (long)blockIdx.x*256 + t; gid < N; gid += (long)gridDim.x*256) {
    int o = (int)(gid & 127);
    int i = (int)((gid >> 7) & 15);
    long b = gid >> 11;
    const float* xb = X + b*16*FIN;
    const float* w0 = W;
    const float* w1 = W + FIN*128;
    float acc = 0.f, h1i = 0.f, h0i = 0.f;
    for (int j = 0; j < 16; ++j) {
      float h1 = 0.f;
      #pragma unroll
      for (int f = 0; f < FIN; ++f) h1 += xb[j*FIN + f] * w1[f*128 + o];
      acc += atts[i*16 + j] * h1;
      if (j == i) h1i = h1;
    }
    #pragma unroll
    for (int f = 0; f < FIN; ++f) h0i += xb[i*FIN + f] * w0[f*128 + o];
    acc += atts[i*16 + i] * (h0i - h1i);
    acc += bias[o];
    Y[gid] = acc;
    s1 += acc; s2 += acc*acc; o_me = o;   // o invariant: stride % 128 == 0
  }
  if (o_me >= 0) { atomicAdd(lsum + o_me, s1); atomicAdd(lsq + o_me, s2); }
  __syncthreads();
  if (t < 128) {
    atomicAdd(stats + t,       lsum[t]);
    atomicAdd(stats + 128 + t, lsq[t]);
  }
}

// ---------------------------------------------------------------------------
// BN finalize: scale/shift from accumulated sum/sumsq.
// ---------------------------------------------------------------------------
__global__ void bnfin(const float* __restrict__ stats, const float* __restrict__ g,
                      const float* __restrict__ beta, float* __restrict__ ss) {
  int t = threadIdx.x;              // 128
  const float invN = 1.f / (8192.f * 16.f);
  float mu  = stats[t] * invN;
  float var = stats[128 + t] * invN - mu*mu;
  float sc  = g[t] * rsqrtf(var + 1e-5f);
  ss[t] = sc; ss[128 + t] = beta[t] - mu*sc;
}

// ---------------------------------------------------------------------------
// out = relu(Y*scale+shift) (+ res);  vectorized float4.
// ---------------------------------------------------------------------------
__global__ __launch_bounds__(256)
void bnrelu(const float4* __restrict__ Y, const float* __restrict__ ss,
            const float4* __restrict__ res, float4* __restrict__ out) {
  const long n4 = (long)8192*16*32;
  for (long gid = (long)blockIdx.x*256 + threadIdx.x; gid < n4;
       gid += (long)gridDim.x*256) {
    int c = (int)(gid & 31) * 4;
    float4 v = Y[gid];
    float4 r;
    r.x = fmaxf(v.x*ss[c+0] + ss[128+c+0], 0.f);
    r.y = fmaxf(v.y*ss[c+1] + ss[128+c+1], 0.f);
    r.z = fmaxf(v.z*ss[c+2] + ss[128+c+2], 0.f);
    r.w = fmaxf(v.w*ss[c+3] + ss[128+c+3], 0.f);
    if (res) { float4 q = res[gid]; r.x += q.x; r.y += q.y; r.z += q.z; r.w += q.w; }
    out[gid] = r;
  }
}

// ---------------------------------------------------------------------------
// Output semgconv: Fin=128, Fout=3. Block = 8 batches.
// ---------------------------------------------------------------------------
__global__ __launch_bounds__(256)
void gconv_out(const float* __restrict__ H, const float* __restrict__ W,
               const float* __restrict__ bias, const float* __restrict__ att,
               float* __restrict__ out) {
  __shared__ float hm[2][8][16][3];
  __shared__ float atts[256];
  int t = threadIdx.x;
  atts[t] = att[t];
  long B0 = (long)blockIdx.x * 8;
  #pragma unroll
  for (int it = 0; it < 3; ++it) {
    int idx = t + it*256;            // 768 dot products of length 128
    int m = idx / 384, rem = idx % 384;
    int b = rem / 48, r2 = rem % 48, j = r2 / 3, o = r2 % 3;
    const float* hr = H + ((B0 + b)*16 + j) * 128;
    const float* wc = W + m*128*3 + o;
    float acc = 0.f;
    for (int f = 0; f < 128; ++f) acc += hr[f] * wc[f*3];
    hm[m][b][j][o] = acc;
  }
  __syncthreads();
  if (t < 384) {
    int b = t / 48, rem = t % 48, i = rem / 3, o = rem % 3;
    float acc = 0.f;
    #pragma unroll
    for (int j = 0; j < 16; ++j) acc += atts[i*16 + j] * hm[1][b][j][o];
    acc += atts[i*16 + i] * (hm[0][b][i][o] - hm[1][b][i][o]);
    out[((B0 + b)*16 + i)*3 + o] = acc + bias[o];
  }
}

// gather merged [B,16,12] from the four branch outputs already in d_out
__global__ __launch_bounds__(256)
void gather_merged(const float* __restrict__ outs, float* __restrict__ merged) {
  const long N = (long)8192*16*12;
  for (long gid = (long)blockIdx.x*256 + threadIdx.x; gid < N;
       gid += (long)gridDim.x*256) {
    int f = (int)(gid % 12); long bj = gid / 12;
    merged[gid] = outs[(long)(1 + f/3)*393216 + bj*3 + (f % 3)];
  }
}

// ---------------------------------------------------------------------------
extern "C" void kernel_launch(void* const* d_in, const int* in_sizes, int n_in,
                              void* d_out, int out_size, void* d_ws, size_t ws_size,
                              hipStream_t stream) {
  const float* x[4]     = {(const float*)d_in[0], (const float*)d_in[1],
                           (const float*)d_in[2], (const float*)d_in[3]};
  const float* W_in     = (const float*)d_in[4];
  const float* b_in     = (const float*)d_in[5];
  const float* e_in     = (const float*)d_in[6];
  const float* g_in     = (const float*)d_in[7];
  const float* beta_in  = (const float*)d_in[8];
  const float* W_cat    = (const float*)d_in[9];
  const float* b_cat    = (const float*)d_in[10];
  const float* e_cat    = (const float*)d_in[11];
  const float* g_cat    = (const float*)d_in[12];
  const float* beta_cat = (const float*)d_in[13];
  const float* W_res    = (const float*)d_in[14];
  const float* b_res    = (const float*)d_in[15];
  const float* e_res    = (const float*)d_in[16];
  const float* g_res    = (const float*)d_in[17];
  const float* beta_res = (const float*)d_in[18];
  const float* W_out    = (const float*)d_in[19];
  const float* b_out    = (const float*)d_in[20];
  const float* e_out    = (const float*)d_in[21];

  char* ws = (char*)d_ws;
  float*     att   = (float*)(ws + 0);                 // 15*256 f32
  _Float16*  Wt    = (_Float16*)(ws + 16384);          // 8*2*128*128 f16
  float*     stats = (float*)(ws + 540672);            // 256 f32
  float*     ssbuf = (float*)(ws + 541696);            // 256 f32
  const size_t ABYTES = (size_t)8192*16*128*4;         // 64 MB
  float* Abuf = (float*)(ws + (1u<<20));
  float* Bbuf = (float*)(ws + (1u<<20) + ABYTES);
  float* Ybuf = (float*)(ws + (1u<<20) + 2*ABYTES);
  float* outp = (float*)d_out;

  const int SMEM = 3*(128*LDSX*2) + (8*2*16*17 + 256 + 128 + 128 + 128)*4; // 124416

  prep_att<<<15, 16, 0, stream>>>(e_in, e_cat, e_res, e_out, att);
  convw<<<1024, 256, 0, stream>>>(W_res, Wt);

  auto res_stack = [&]() {
    for (int blk = 0; blk < 4; ++blk) {
      for (int sub = 0; sub < 2; ++sub) {
        int l = 2*blk + sub;
        zero_stats<<<1, 256, 0, stream>>>(stats);
        gconv128<<<1024, 256, SMEM, stream>>>(
            sub ? Bbuf : Abuf,
            Wt + (l*2 + 0)*16384, Wt + (l*2 + 1)*16384,
            b_res + l*128, att + (2 + l)*256, Ybuf, stats);
        bnfin<<<1, 128, 0, stream>>>(stats, g_res + l*128, beta_res + l*128, ssbuf);
        bnrelu<<<2048, 256, 0, stream>>>((const float4*)Ybuf, ssbuf,
                                         sub ? (const float4*)Abuf : nullptr,
                                         (float4*)(sub ? Abuf : Bbuf));
      }
    }
  };

  // four input branches (shared in-block + res-stack weights)
  for (int br = 0; br < 4; ++br) {
    zero_stats<<<1, 256, 0, stream>>>(stats);
    gconv_in<2><<<2048, 256, 0, stream>>>(x[br], W_in, b_in, att + 0*256, Ybuf, stats);
    bnfin<<<1, 128, 0, stream>>>(stats, g_in, beta_in, ssbuf);
    bnrelu<<<2048, 256, 0, stream>>>((const float4*)Ybuf, ssbuf, nullptr, (float4*)Abuf);
    res_stack();
    gconv_out<<<1024, 256, 0, stream>>>(Abuf, W_out + br*2*128*3, b_out + br*3,
                                        att + (10 + br)*256,
                                        outp + (long)(1 + br)*393216);
  }

  // concat branch
  gather_merged<<<1024, 256, 0, stream>>>(outp, Bbuf);   // Bbuf free here
  zero_stats<<<1, 256, 0, stream>>>(stats);
  gconv_in<12><<<2048, 256, 0, stream>>>(Bbuf, W_cat, b_cat, att + 1*256, Ybuf, stats);
  bnfin<<<1, 128, 0, stream>>>(stats, g_cat, beta_cat, ssbuf);
  bnrelu<<<2048, 256, 0, stream>>>((const float4*)Ybuf, ssbuf, nullptr, (float4*)Abuf);
  res_stack();
  gconv_out<<<1024, 256, 0, stream>>>(Abuf, W_out + 4*2*128*3, b_out + 4*3,
                                      att + 14*256, outp);
  (void)in_sizes; (void)n_in; (void)out_size; (void)ws_size;
}
